// SparseAutoencoder_39195871543502
// MI455X (gfx1250) — matrix operand order repaired and checked by reading, not compiled, not run on hardware
//
#include <hip/hip_runtime.h>
#include <hip/hip_bf16.h>
#include <cstdint>

#define B_N 4096
#define D_N 4096
#define M_N 32768
#define K_ACT 64
#define MULTIK 192
#define AUXK 128
#define DEADTH 256
#define KTOTAL (K_ACT * B_N)   // 262144
#define SELT 512

typedef __attribute__((ext_vector_type(16))) __bf16 bf16x16;
typedef __attribute__((ext_vector_type(8)))  float  f32x8;

union Frag { bf16x16 bf; unsigned w[8]; uint4 q[2]; };

__device__ __forceinline__ unsigned bfround(unsigned u) {
  return u + 0x7FFFu + ((u >> 16) & 1u);            // round-to-nearest-even
}
__device__ __forceinline__ unsigned short f2bf(float x) {
  return (unsigned short)(bfround(__float_as_uint(x)) >> 16);
}
// pack two f32 -> packed bf16x2 word: one v_perm_b32 merges the high halves
__device__ __forceinline__ unsigned pk2(float lo, float hi) {
  unsigned rl = bfround(__float_as_uint(lo));
  unsigned rh = bfround(__float_as_uint(hi));
  return __builtin_amdgcn_perm(rh, rl, 0x07060302u); // {rh[3],rh[2],rl[3],rl[2]}
}

// A fragment (16 rows x 32 K bf16): per lane elems 0..7 = K[h*8..], 8..15 = K[16+h*8..]
__device__ __forceinline__ bf16x16 fragA_bf16(const unsigned short* __restrict__ p) {
  Frag f;
  f.q[0] = *(const uint4*)(p);
  f.q[1] = *(const uint4*)(p + 16);
  return f.bf;
}
// B fragment (32 K x 16 cols bf16): per lane elems 0..7 = K[h*8..], 8..15 = K[16+h*8..]
__device__ __forceinline__ bf16x16 fragB_bf16(const unsigned short* __restrict__ p) {
  Frag f;
  f.q[0] = *(const uint4*)(p);
  f.q[1] = *(const uint4*)(p + 16);
  return f.bf;
}

// ---------------------------------------------------------------------------
// Encoder GEMM, all-bf16: pre_act[B,M] = xb @ Wenc_bf16^T + neuron_bias
// 8 waves (2 batch x 4 neuron); wave tile 64(batch) x 32(neuron).
// blockIdx.x = batch (fast) so concurrent WGs share W_enc tiles via 192MB L2.
// ---------------------------------------------------------------------------
__global__ __launch_bounds__(256) void sae_encode(const unsigned short* __restrict__ xb,
    const unsigned short* __restrict__ wb, const float* __restrict__ nbias,
    float* __restrict__ preact) {
  const int lane = threadIdx.x & 31, w = threadIdx.x >> 5;
  const int bBase = blockIdx.x * 128 + (w & 1) * 64;
  const int nBase = blockIdx.y * 128 + (w >> 1) * 32;
  const int r = lane & 15, h = lane >> 4;
  const unsigned short* pa[4];
#pragma unroll
  for (int i = 0; i < 4; ++i) pa[i] = xb + (size_t)(bBase + 16 * i + r) * D_N + h * 8;
  const unsigned short* pb[2];
#pragma unroll
  for (int j = 0; j < 2; ++j) pb[j] = wb + (size_t)(nBase + 16 * j + r) * D_N + h * 8;

  f32x8 acc[4][2] = {};
  for (int d0 = 0; d0 < D_N; d0 += 32) {
#pragma unroll
    for (int j = 0; j < 2; ++j) __builtin_prefetch(pb[j] + d0 + 256, 0, 0);
    bf16x16 bfrag[2];
#pragma unroll
    for (int j = 0; j < 2; ++j) bfrag[j] = fragB_bf16(pb[j] + d0);
#pragma unroll
    for (int i = 0; i < 4; ++i) {
      bf16x16 afrag = fragA_bf16(pa[i] + d0);
#pragma unroll
      for (int j = 0; j < 2; ++j)
        acc[i][j] = __builtin_amdgcn_wmma_f32_16x16x32_bf16(
            false, afrag, false, bfrag[j], (short)0, acc[i][j], false, false);
    }
  }
  const int nl = lane & 15;
#pragma unroll
  for (int i = 0; i < 4; ++i)
#pragma unroll
    for (int j = 0; j < 2; ++j) {
      const int ncol = nBase + 16 * j + nl;
      const float nb = nbias[ncol];
#pragma unroll
      for (int rr = 0; rr < 8; ++rr) {
        const int brow = bBase + 16 * i + rr + 8 * h;
        preact[(size_t)brow * M_N + ncol] = acc[i][j][rr] + nb;
      }
    }
}

// ---------------------------------------------------------------------------
// Fused decoder GEMM, all-bf16: recon = activ_bf16 @ Wdec_bf16^T + b;
// mrecon = multik_bf16 @ Wdec_bf16^T + b. W_dec fragment feeds both.
// 8 waves (2 batch x 4 d); wave tile 32(batch) x 32(d).
// ---------------------------------------------------------------------------
__global__ __launch_bounds__(256) void sae_decode(const unsigned short* __restrict__ abf,
    const unsigned short* __restrict__ mk, const unsigned short* __restrict__ wdb,
    const float* __restrict__ ibias, float* __restrict__ recon, float* __restrict__ mrecon) {
  const int lane = threadIdx.x & 31, w = threadIdx.x >> 5;
  const int bBase = blockIdx.x * 64 + (w & 1) * 32;
  const int dBase = blockIdx.y * 128 + (w >> 1) * 32;
  const int r = lane & 15, h = lane >> 4;
  const unsigned short* p1[2];
  const unsigned short* p2[2];
#pragma unroll
  for (int i = 0; i < 2; ++i) {
    p1[i] = abf + (size_t)(bBase + 16 * i + r) * M_N + h * 8;
    p2[i] = mk  + (size_t)(bBase + 16 * i + r) * M_N + h * 8;
  }
  const unsigned short* pb[2];
#pragma unroll
  for (int j = 0; j < 2; ++j) pb[j] = wdb + (size_t)(dBase + 16 * j + r) * M_N + h * 8;

  f32x8 acc1[2][2] = {}, acc2[2][2] = {};
  for (int m0 = 0; m0 < M_N; m0 += 32) {
#pragma unroll
    for (int j = 0; j < 2; ++j) __builtin_prefetch(pb[j] + m0 + 256, 0, 0);
    bf16x16 bfrag[2];
#pragma unroll
    for (int j = 0; j < 2; ++j) bfrag[j] = fragB_bf16(pb[j] + m0);
#pragma unroll
    for (int i = 0; i < 2; ++i) {
      bf16x16 a1 = fragA_bf16(p1[i] + m0);
      bf16x16 a2 = fragA_bf16(p2[i] + m0);
#pragma unroll
      for (int j = 0; j < 2; ++j) {
        acc1[i][j] = __builtin_amdgcn_wmma_f32_16x16x32_bf16(
            false, a1, false, bfrag[j], (short)0, acc1[i][j], false, false);
        acc2[i][j] = __builtin_amdgcn_wmma_f32_16x16x32_bf16(
            false, a2, false, bfrag[j], (short)0, acc2[i][j], false, false);
      }
    }
  }
  const int nl = lane & 15;
#pragma unroll
  for (int i = 0; i < 2; ++i)
#pragma unroll
    for (int j = 0; j < 2; ++j) {
      const int dcol = dBase + 16 * j + nl;
      const float ib = ibias[dcol];
#pragma unroll
      for (int rr = 0; rr < 8; ++rr) {
        const int brow = bBase + 16 * i + rr + 8 * h;
        recon [(size_t)brow * D_N + dcol] = acc1[i][j][rr] + ib;
        mrecon[(size_t)brow * D_N + dcol] = acc2[i][j][rr] + ib;
      }
    }
}

// --------------------- generic fp32 -> bf16 convert ------------------------
__global__ __launch_bounds__(256) void sae_cvt(const float* __restrict__ src,
                                               unsigned short* __restrict__ dst,
                                               unsigned long long N) {
  const size_t stride = (size_t)gridDim.x * blockDim.x * 8;
  for (size_t i = ((size_t)blockIdx.x * blockDim.x + threadIdx.x) * 8; i < N; i += stride) {
    float4 a = *(const float4*)(src + i);
    float4 b = *(const float4*)(src + i + 4);
    uint4 o;
    o.x = pk2(a.x, a.y); o.y = pk2(a.z, a.w);
    o.z = pk2(b.x, b.y); o.w = pk2(b.z, b.w);
    *(uint4*)(dst + i) = o;
  }
}

// x_bf16 = bf16(x - input_bias), row length D_N (power of two)
__global__ __launch_bounds__(256) void sae_cvt_x(const float* __restrict__ x,
                                                 const float* __restrict__ ibias,
                                                 unsigned short* __restrict__ xb) {
  const size_t N = (size_t)B_N * D_N;
  const size_t stride = (size_t)gridDim.x * blockDim.x * 8;
  for (size_t i = ((size_t)blockIdx.x * blockDim.x + threadIdx.x) * 8; i < N; i += stride) {
    const size_t d = i & (size_t)(D_N - 1);
    float4 a = *(const float4*)(x + i);
    float4 b = *(const float4*)(x + i + 4);
    float4 c = *(const float4*)(ibias + d);
    float4 e = *(const float4*)(ibias + d + 4);
    uint4 o;
    o.x = pk2(a.x - c.x, a.y - c.y); o.y = pk2(a.z - c.z, a.w - c.w);
    o.z = pk2(b.x - e.x, b.y - e.y); o.w = pk2(b.z - e.z, b.w - e.w);
    *(uint4*)(xb + i) = o;
  }
}

// ---------------------------------------------------------------------------
// Per-row top-192 by |pre_act| -> dense bf16 multik row. Deterministic
// 11-bit radix threshold + index-order tie ranks; folds row histogram into
// the global round-A histogram for the batch top-k.
// ---------------------------------------------------------------------------
__global__ __launch_bounds__(SELT) void sae_multik(const float* __restrict__ preact,
    unsigned short* __restrict__ mk, unsigned* __restrict__ gHistA) {
  __shared__ unsigned hist[2048];
  __shared__ unsigned scn[SELT];
  __shared__ int sThr, sRem;
  const int tid = threadIdx.x;
  const size_t b = blockIdx.x;
  const float* row = preact + b * (size_t)M_N;
  for (int i = tid; i < 2048; i += SELT) hist[i] = 0;
  __syncthreads();
  const int CH = M_N / SELT;                 // 64 contiguous elems per thread
  const int i0 = tid * CH;
  for (int i = 0; i < CH; ++i) {
    unsigned key = __float_as_uint(row[i0 + i]) & 0x7FFFFFFFu;
    atomicAdd(&hist[key >> 20], 1u);
  }
  __syncthreads();
  for (int i = tid; i < 2048; i += SELT) if (hist[i]) atomicAdd(&gHistA[i], hist[i]);
  if (tid == 0) {
    unsigned acc = 0; int thr = 0; int rem = MULTIK;
    for (int bin = 2047; bin >= 0; --bin) {
      unsigned c = hist[bin];
      if (acc + c >= (unsigned)MULTIK) { thr = bin; rem = MULTIK - (int)acc; break; }
      acc += c;
    }
    sThr = thr; sRem = rem;
  }
  __syncthreads();
  const int thr = sThr, rem = sRem;
  unsigned cnt = 0;
  for (int i = 0; i < CH; ++i) {
    unsigned key = __float_as_uint(row[i0 + i]) & 0x7FFFFFFFu;
    cnt += (unsigned)((int)(key >> 20) == thr);
  }
  scn[tid] = cnt;
  __syncthreads();
  for (int off = 1; off < SELT; off <<= 1) {   // Hillis-Steele inclusive scan
    unsigned v = (tid >= off) ? scn[tid - off] : 0u;
    __syncthreads();
    scn[tid] += v;
    __syncthreads();
  }
  int rank = (int)(scn[tid] - cnt);            // exclusive tie rank
  unsigned short* out = mk + b * (size_t)M_N;
  for (int i = 0; i < CH; ++i) {
    float v = row[i0 + i];
    unsigned key = __float_as_uint(v) & 0x7FFFFFFFu;
    int bin = (int)(key >> 20);
    bool sel = (bin > thr) || (bin == thr && (rank++ < rem));
    out[i0 + i] = sel ? f2bf(v) : (unsigned short)0;
  }
}

// ---------------------------------------------------------------------------
// Per-row top-128 of dead-masked signed pre_act -> aux_vals / aux_idx.
// ---------------------------------------------------------------------------
__global__ __launch_bounds__(SELT) void sae_aux(const float* __restrict__ preact,
    const int* __restrict__ steps, float* __restrict__ avals, float* __restrict__ aidx) {
  __shared__ unsigned hist[2048];
  __shared__ unsigned scn[SELT];
  __shared__ int sThr, sRem;
  const int tid = threadIdx.x;
  const size_t b = blockIdx.x;
  const float* row = preact + b * (size_t)M_N;
  for (int i = tid; i < 2048; i += SELT) hist[i] = 0;
  __syncthreads();
  const int CH = M_N / SELT;
  const int i0 = tid * CH;
  for (int i = 0; i < CH; ++i) {
    const int m = i0 + i;
    float v = (steps[m] > DEADTH) ? row[m] : 0.0f;
    unsigned u = __float_as_uint(v);
    unsigned key = u ^ ((u & 0x80000000u) ? 0xFFFFFFFFu : 0x80000000u);
    atomicAdd(&hist[key >> 21], 1u);
  }
  __syncthreads();
  if (tid == 0) {
    unsigned acc = 0; int thr = 0; int rem = AUXK;
    for (int bin = 2047; bin >= 0; --bin) {
      unsigned c = hist[bin];
      if (acc + c >= (unsigned)AUXK) { thr = bin; rem = AUXK - (int)acc; break; }
      acc += c;
    }
    sThr = thr; sRem = rem;
  }
  __syncthreads();
  const int thr = sThr, rem = sRem;
  unsigned above = 0, ties = 0;
  for (int i = 0; i < CH; ++i) {
    const int m = i0 + i;
    float v = (steps[m] > DEADTH) ? row[m] : 0.0f;
    unsigned u = __float_as_uint(v);
    unsigned key = u ^ ((u & 0x80000000u) ? 0xFFFFFFFFu : 0x80000000u);
    int bin = (int)(key >> 21);
    above += (unsigned)(bin > thr);
    ties  += (unsigned)(bin == thr);
  }
  const unsigned mine = (above << 16) | ties;
  scn[tid] = mine;
  __syncthreads();
  for (int off = 1; off < SELT; off <<= 1) {
    unsigned v = (tid >= off) ? scn[tid - off] : 0u;
    __syncthreads();
    scn[tid] += v;
    __syncthreads();
  }
  const unsigned excl = scn[tid] - mine;
  unsigned runA = excl >> 16;
  unsigned runT = excl & 0xFFFFu;
  for (int i = 0; i < CH; ++i) {
    const int m = i0 + i;
    float v = (steps[m] > DEADTH) ? row[m] : 0.0f;
    unsigned u = __float_as_uint(v);
    unsigned key = u ^ ((u & 0x80000000u) ? 0xFFFFFFFFu : 0x80000000u);
    int bin = (int)(key >> 21);
    if (bin > thr) {
      unsigned t = runT < (unsigned)rem ? runT : (unsigned)rem;
      unsigned pos = runA + t;
      avals[b * AUXK + pos] = v;
      aidx [b * AUXK + pos] = (float)m;
      ++runA;
    } else if (bin == thr) {
      if (runT < (unsigned)rem) {
        unsigned pos = runA + runT;
        avals[b * AUXK + pos] = v;
        aidx [b * AUXK + pos] = (float)m;
      }
      ++runT;
    }
  }
}

// --------------------- global batch top-k threshold passes -----------------
__global__ void sae_scanA(const unsigned* __restrict__ gHistA, unsigned* __restrict__ ctrl) {
  if (threadIdx.x == 0) {
    unsigned acc = 0, thr = 0, rem = KTOTAL;
    for (int bin = 2047; bin >= 0; --bin) {
      unsigned c = gHistA[bin];
      if (acc + c >= (unsigned)KTOTAL) { thr = (unsigned)bin; rem = KTOTAL - acc; break; }
      acc += c;
    }
    ctrl[0] = thr; ctrl[1] = rem;
  }
}

__global__ __launch_bounds__(256) void sae_histB(const float* __restrict__ preact,
    const unsigned* __restrict__ ctrl, unsigned* __restrict__ gHistB) {
  __shared__ unsigned hist[2048];
  for (int i = threadIdx.x; i < 2048; i += 256) hist[i] = 0;
  __syncthreads();
  const unsigned thrA = ctrl[0];
  const size_t N = (size_t)B_N * M_N;
  const size_t stride = (size_t)gridDim.x * blockDim.x;
  for (size_t i = (size_t)blockIdx.x * blockDim.x + threadIdx.x; i < N; i += stride) {
    unsigned key = __float_as_uint(preact[i]) & 0x7FFFFFFFu;
    if ((key >> 20) == thrA) atomicAdd(&hist[(key >> 9) & 0x7FFu], 1u);
  }
  __syncthreads();
  for (int i = threadIdx.x; i < 2048; i += 256) if (hist[i]) atomicAdd(&gHistB[i], hist[i]);
}

__global__ void sae_scanB(const unsigned* __restrict__ gHistB, unsigned* __restrict__ ctrl) {
  if (threadIdx.x == 0) {
    const unsigned rem = ctrl[1];
    unsigned acc = 0, thrB = 0;
    for (int bin = 2047; bin >= 0; --bin) {
      unsigned c = gHistB[bin];
      if (acc + c >= rem) { thrB = (unsigned)bin; break; }
      acc += c;
    }
    ctrl[2] = (ctrl[0] << 11) | thrB;          // 22-bit threshold key
  }
}

// Mask pre_act -> activ (fp32, in place) AND emit bf16 copy for the decoder.
__global__ __launch_bounds__(256) void sae_mask(float* __restrict__ activ,
    unsigned short* __restrict__ abf, const unsigned* __restrict__ ctrl) {
  const unsigned thr22 = ctrl[2];
  const size_t N = (size_t)B_N * M_N;
  const size_t stride = (size_t)gridDim.x * blockDim.x * 4;
  for (size_t i = ((size_t)blockIdx.x * blockDim.x + threadIdx.x) * 4; i < N; i += stride) {
    float4 v = *(const float4*)(activ + i);
    float vv[4] = {v.x, v.y, v.z, v.w};
#pragma unroll
    for (int j = 0; j < 4; ++j) {
      unsigned key = __float_as_uint(vv[j]) & 0x7FFFFFFFu;
      if ((key >> 9) < thr22) vv[j] = 0.0f;    // keep >= threshold key
    }
    float4 o; o.x = vv[0]; o.y = vv[1]; o.z = vv[2]; o.w = vv[3];
    *(float4*)(activ + i) = o;
    uint2 p; p.x = pk2(vv[0], vv[1]); p.y = pk2(vv[2], vv[3]);
    *(uint2*)(abf + i) = p;
  }
}

// ---------------------------------------------------------------------------
extern "C" void kernel_launch(void* const* d_in, const int* in_sizes, int n_in,
                              void* d_out, int out_size, void* d_ws, size_t ws_size,
                              hipStream_t stream) {
  (void)in_sizes; (void)n_in; (void)out_size; (void)ws_size;
  const float* x     = (const float*)d_in[0];
  const float* Wenc  = (const float*)d_in[1];
  const float* Wdec  = (const float*)d_in[2];
  const float* ibias = (const float*)d_in[3];
  const float* nbias = (const float*)d_in[4];
  const int*   steps = (const int*)d_in[5];

  // d_out: [recon B*D][activ B*M][multik_recon B*D][aux_vals B*128][aux_idx B*128]
  float* recon  = (float*)d_out;
  float* activ  = recon  + (size_t)B_N * D_N;    // holds pre_act until masked
  float* mrecon = activ  + (size_t)B_N * M_N;
  float* avals  = mrecon + (size_t)B_N * D_N;
  float* aidx   = avals  + (size_t)B_N * AUXK;

  // ws: [multik bf16 B*M][activ bf16 B*M][weight bf16 M*D (Wenc then Wdec)]
  //     [x bf16 B*D][histA][histB][ctrl]
  unsigned short* mk  = (unsigned short*)d_ws;
  unsigned short* abf = mk  + (size_t)B_N * M_N;
  unsigned short* wb  = abf + (size_t)B_N * M_N;   // shared Wenc/Wdec bf16 slot
  unsigned short* xb  = wb  + (size_t)M_N * D_N;
  unsigned* gHistA = (unsigned*)(xb + (size_t)B_N * D_N);
  unsigned* gHistB = gHistA + 2048;
  unsigned* ctrl   = gHistB + 2048;

  const unsigned long long NW = (unsigned long long)M_N * D_N;

  hipMemsetAsync(gHistA, 0, (2048 + 2048 + 16) * sizeof(unsigned), stream);

  sae_cvt_x<<<dim3(4096), 256, 0, stream>>>(x, ibias, xb);
  sae_cvt  <<<dim3(8192), 256, 0, stream>>>(Wenc, wb, NW);
  sae_encode<<<dim3(B_N / 128, M_N / 128), 256, 0, stream>>>(xb, wb, nbias, activ);
  sae_multik<<<dim3(B_N), SELT, 0, stream>>>(activ, mk, gHistA);
  sae_aux   <<<dim3(B_N), SELT, 0, stream>>>(activ, steps, avals, aidx);
  sae_scanA <<<dim3(1), 32, 0, stream>>>(gHistA, ctrl);
  sae_histB <<<dim3(8192), 256, 0, stream>>>(activ, ctrl, gHistB);
  sae_scanB <<<dim3(1), 32, 0, stream>>>(gHistB, ctrl);
  sae_mask  <<<dim3(16384), 256, 0, stream>>>(activ, abf, ctrl);
  sae_cvt   <<<dim3(8192), 256, 0, stream>>>(Wdec, wb, NW);   // reuse weight slot
  sae_decode<<<dim3(B_N / 64, D_N / 128), 256, 0, stream>>>(abf, mk, wb, ibias, recon, mrecon);
}
